// mylstm_69123203662583
// MI455X (gfx1250) — matrix operand order repaired and checked by reading, not compiled, not run on hardware
//
#include <hip/hip_runtime.h>
#include <hip/hip_bf16.h>

// ---------------- problem constants ----------------
#define B_TOT   65536
#define T_STEPS 20
#define I_DIM   36
#define H_DIM   30
#define GATES   120          // 4*H
#define NTILES  8            // 120 gate cols padded to 8 tiles of 16

#define BLOCK_THREADS 128    // 4 waves (wave32)
#define WAVES         4
#define ROWS_PER_BLOCK 64    // 4 waves * 16 batch rows

typedef __attribute__((ext_vector_type(16))) _Float16 v16h;
typedef __attribute__((ext_vector_type(8)))  _Float16 v8h;
typedef __attribute__((ext_vector_type(8)))  float    v8f;
typedef __attribute__((ext_vector_type(4)))  float    f4;

// ---------------- LDS layout (bytes) ----------------
// 24 B-fragments (8 tiles x 3 k-slots), each: 32 lanes * 32B, swizzled
#define WF_FRAG_BYTES 1088                      // 31*32 + 3*16 + 32 = 1072, pad to 1088
#define WF_OFF   0
#define WF_BYTES (24 * WF_FRAG_BYTES)           // 26112
#define BIAS_OFF WF_BYTES                       // 128 floats = 512B
#define H_OFF    (BIAS_OFF + 512)               // 26624
#define H_ROW_HALFS 40                          // 30 used + pad (k=30..31 zeroed)
#define H_ROW_BYTES (H_ROW_HALFS * 2)           // 80
#define H_BYTES  (ROWS_PER_BLOCK * H_ROW_BYTES) // 5120
#define G_OFF    (H_OFF + H_BYTES)              // 31744
#define G_STRIDE 130                            // floats; 8*130 % 64 = 16 -> conflict-free stores
#define G_BYTES  (ROWS_PER_BLOCK * G_STRIDE * 4)// 33280
#define LDS_TOTAL (G_OFF + G_BYTES)             // 65024 bytes (< 64KB, safe everywhere)

// ---------------- activations ----------------
__device__ __forceinline__ float tanh_fast(float x) {
#if __has_builtin(__builtin_amdgcn_tanhf)
  return __builtin_amdgcn_tanhf(x);          // v_tanh_f32 (gfx1250 TRANS op)
#else
  float ax = fabsf(x);
  float e  = __expf(-2.0f * ax);
  float r  = (1.0f - e) / (1.0f + e);
  return copysignf(r, x);
#endif
}
__device__ __forceinline__ float sigmoid_fast(float x) {
  return 0.5f + 0.5f * tanh_fast(0.5f * x);
}

__global__ __launch_bounds__(BLOCK_THREADS)
void lstm_fused_kernel(const float* __restrict__ x,
                       const float* __restrict__ Wih,
                       const float* __restrict__ Whh,
                       const float* __restrict__ bih,
                       const float* __restrict__ bhh,
                       float* __restrict__ out) {
  __shared__ __align__(16) unsigned char smem[LDS_TOTAL];
  _Float16* wf     = (_Float16*)(smem + WF_OFF);
  float*    bias_s = (float*)(smem + BIAS_OFF);
  _Float16* h_s    = (_Float16*)(smem + H_OFF);
  float*    g_s    = (float*)(smem + G_OFF);

  const int tid = threadIdx.x;

  // ---- one-time: stage W into exact WMMA B-fragment layout (f16), swizzled ----
  // B-matrix 32x16 (KxN): lane = N col (lanes 0-15 K=0-7,16-23; lanes 16-31 K=8-15,24-31),
  // element e within v16h -> K = (lane/16)*8 + e for e<8, 16 + (lane/16)*8 + (e-8) for e>=8.
  for (int idx = tid; idx < 24 * 32; idx += BLOCK_THREADS) {
    const int f = idx >> 5, l = idx & 31;
    const int j = f / 3, s = f - 3 * j;      // tile, k-slot
    const int n = j * 16 + (l & 15);         // gate column
    const int kbase = (l >> 4) * 8;
    _Float16 tmp[16];
#pragma unroll
    for (int e = 0; e < 16; ++e) {
      const int K = (e < 8) ? (kbase + e) : (16 + kbase + (e - 8));
      float v = 0.0f;
      if (n < GATES) {
        if (s == 0)       v = Wih[n * I_DIM + K];                       // x, K = 0..31 (<36)
        else if (s == 1)  { int kk = 32 + K; if (kk < I_DIM) v = Wih[n * I_DIM + kk]; }
        else              { if (K < H_DIM) v = Whh[n * H_DIM + K]; }    // h, K = 0..29
      }
      tmp[e] = (_Float16)v;
    }
    _Float16* dst = wf + ((f * WF_FRAG_BYTES + l * 32 + (l >> 3) * 16) >> 1);
#pragma unroll
    for (int e = 0; e < 16; ++e) dst[e] = tmp[e];
  }
  if (tid < 128) bias_s[tid] = (tid < GATES) ? (bih[tid] + bhh[tid]) : 0.0f;
  for (int i2 = tid; i2 < ROWS_PER_BLOCK * H_ROW_HALFS; i2 += BLOCK_THREADS)
    h_s[i2] = (_Float16)0.0f;                // h0 = 0, incl. K=30..31 padding
  __syncthreads();
  // ---- after this point: waves are fully independent (no more barriers) ----

  const int wave = tid >> 5, lane = tid & 31;
  const int mloc = lane & 15, hi = lane >> 4, kb = hi * 8;
  const int rowLoc = wave * 16 + mloc;                       // 0..63
  const size_t batch = (size_t)blockIdx.x * ROWS_PER_BLOCK + rowLoc;
  const float* px = x + batch * (T_STEPS * I_DIM);           // row stride 144B (16B aligned)
  const int laneoff = lane * 32 + (lane >> 3) * 16;          // swizzled B-frag lane offset

  const v8h* hp0 = (const v8h*)((unsigned char*)h_s + rowLoc * H_ROW_BYTES + kb * 2);
  const v8h* hp1 = (const v8h*)((unsigned char*)h_s + rowLoc * H_ROW_BYTES + 32 + kb * 2);

  float c_reg[15], h_reg[15];
#pragma unroll
  for (int p = 0; p < 15; ++p) { c_reg[p] = 0.0f; h_reg[p] = 0.0f; }

  float btile[NTILES];
#pragma unroll
  for (int j = 0; j < NTILES; ++j) btile[j] = bias_s[j * 16 + mloc];

  for (int t = 0; t < T_STEPS; ++t) {
    const float* pxt = px + t * I_DIM;
    // ---- A fragment (x_t, 16x32 f16): lane half hi=0 -> K 0-7 & 16-23; hi=1 -> K 8-15 & 24-31
    f4 f0a = *(const f4*)(pxt + kb);
    f4 f0b = *(const f4*)(pxt + kb + 4);
    f4 f0c = *(const f4*)(pxt + 16 + kb);
    f4 f0d = *(const f4*)(pxt + 20 + kb);
    f4 f1  = {0.0f, 0.0f, 0.0f, 0.0f};
    if (!hi) f1 = *(const f4*)(pxt + 32);    // K=32..35 only; lanes 16-31 stay zero
    v16h a0, a1;
    {
      float tv[16] = {f0a.x, f0a.y, f0a.z, f0a.w, f0b.x, f0b.y, f0b.z, f0b.w,
                      f0c.x, f0c.y, f0c.z, f0c.w, f0d.x, f0d.y, f0d.z, f0d.w};
#pragma unroll
      for (int e = 0; e < 16; ++e) a0[e] = (_Float16)tv[e];
    }
    a1[0] = (_Float16)f1.x; a1[1] = (_Float16)f1.y;
    a1[2] = (_Float16)f1.z; a1[3] = (_Float16)f1.w;
#pragma unroll
    for (int e = 4; e < 16; ++e) a1[e] = (_Float16)0.0f;
    // ---- A fragment (h_{t-1}) from LDS (K=30..31 are stored zeros)
    v8h hlo = *hp0, hhi = *hp1;
    v16h ah;
#pragma unroll
    for (int e = 0; e < 8; ++e) { ah[e] = hlo[e]; ah[8 + e] = hhi[e]; }

    // ---- gates = [x|h] @ [Wih|Whh]^T + b : 8 N-tiles x 3 WMMAs ----
    const unsigned char* wb = (const unsigned char*)wf;
#pragma unroll
    for (int j = 0; j < NTILES; ++j) {
      const float bsv = btile[j];
      v8f acc = {bsv, bsv, bsv, bsv, bsv, bsv, bsv, bsv};
      v16h b0, b1, b2;
      {
        v8h lo = *(const v8h*)(wb + (j * 3 + 0) * WF_FRAG_BYTES + laneoff);
        v8h hi2 = *(const v8h*)(wb + (j * 3 + 0) * WF_FRAG_BYTES + laneoff + 16);
#pragma unroll
        for (int e = 0; e < 8; ++e) { b0[e] = lo[e]; b0[8 + e] = hi2[e]; }
      }
      {
        v8h lo = *(const v8h*)(wb + (j * 3 + 1) * WF_FRAG_BYTES + laneoff);
        v8h hi2 = *(const v8h*)(wb + (j * 3 + 1) * WF_FRAG_BYTES + laneoff + 16);
#pragma unroll
        for (int e = 0; e < 8; ++e) { b1[e] = lo[e]; b1[8 + e] = hi2[e]; }
      }
      {
        v8h lo = *(const v8h*)(wb + (j * 3 + 2) * WF_FRAG_BYTES + laneoff);
        v8h hi2 = *(const v8h*)(wb + (j * 3 + 2) * WF_FRAG_BYTES + laneoff + 16);
#pragma unroll
        for (int e = 0; e < 8; ++e) { b2[e] = lo[e]; b2[8 + e] = hi2[e]; }
      }
      acc = __builtin_amdgcn_wmma_f32_16x16x32_f16(false, a0, false, b0, (short)0, acc, false, false);
      acc = __builtin_amdgcn_wmma_f32_16x16x32_f16(false, a1, false, b1, (short)0, acc, false, false);
      acc = __builtin_amdgcn_wmma_f32_16x16x32_f16(false, ah, false, b2, (short)0, acc, false, false);
      // C/D layout: VGPR r, lane -> row M = r + 8*hi, col N = j*16 + mloc
      float* gr = g_s + (wave * 16 + hi * 8) * G_STRIDE + j * 16 + mloc;
#pragma unroll
      for (int r = 0; r < 8; ++r) gr[r * G_STRIDE] = acc[r];
    }

    // ---- elementwise LSTM cell: 16 rows x 30 cols = 480 cells / wave, 15 per lane ----
#pragma unroll
    for (int p = 0; p < 15; ++p) {
      const int q = p * 32 + lane;
      const int m = q / 30, n = q - m * 30;
      const float* gp = g_s + (wave * 16 + m) * G_STRIDE + n;
      const float it = sigmoid_fast(gp[0]);
      const float ft = sigmoid_fast(gp[30]);
      const float gt = tanh_fast(gp[60]);
      const float ot = sigmoid_fast(gp[90]);
      const float cn = ft * c_reg[p] + it * gt;
      c_reg[p] = cn;
      const float hn = ot * tanh_fast(cn);
      h_reg[p] = hn;
      h_s[(wave * 16 + m) * H_ROW_HALFS + n] = (_Float16)hn;   // for next step's A(h)
      const size_t bq = (size_t)blockIdx.x * ROWS_PER_BLOCK + wave * 16 + m;
      out[bq * (size_t)(T_STEPS * H_DIM) + (size_t)t * H_DIM + n] = hn;
    }
  }

  // ---- final h_n, c_n (concatenated after out) ----
  float* hn_out = out + (size_t)B_TOT * T_STEPS * H_DIM;
  float* cn_out = hn_out + (size_t)B_TOT * H_DIM;
#pragma unroll
  for (int p = 0; p < 15; ++p) {
    const int q = p * 32 + lane;
    const int m = q / 30, n = q - m * 30;
    const size_t bq = (size_t)blockIdx.x * ROWS_PER_BLOCK + wave * 16 + m;
    hn_out[bq * H_DIM + n] = h_reg[p];
    cn_out[bq * H_DIM + n] = c_reg[p];
  }
}

extern "C" void kernel_launch(void* const* d_in, const int* in_sizes, int n_in,
                              void* d_out, int out_size, void* d_ws, size_t ws_size,
                              hipStream_t stream) {
  const float* x   = (const float*)d_in[0];
  const float* Wih = (const float*)d_in[1];
  const float* Whh = (const float*)d_in[2];
  const float* bih = (const float*)d_in[3];
  const float* bhh = (const float*)d_in[4];
  float* out = (float*)d_out;
  (void)in_sizes; (void)n_in; (void)out_size; (void)d_ws; (void)ws_size;

  dim3 grid(B_TOT / ROWS_PER_BLOCK);   // 1024 blocks
  dim3 block(BLOCK_THREADS);           // 128 threads = 4 waves
  lstm_fused_kernel<<<grid, block, 0, stream>>>(x, Wih, Whh, bih, bhh, out);
}
